// TransformerDecoderLayer_77232101917204
// MI455X (gfx1250) — matrix operand order repaired and checked by reading, not compiled, not run on hardware
//
#include <hip/hip_runtime.h>

// ---------------------------------------------------------------------------
// Transformer decoder layer for MI455X (gfx1250, wave32, WMMA).
// B=4, S=1024, D=1024, H=16, DH=64, FF=4096.
// All GEMMs: bf16 inputs, f32 accumulate via v_wmma_f32_16x16x32_bf16.
// Global->LDS staging uses GLOBAL_LOAD_ASYNC_TO_LDS_B128 (ASYNCcnt) with a
// double-buffered K pipeline when the builtin is available.
// Workspace budget ~152 MB (attn lives directly in d_out).
// ---------------------------------------------------------------------------

typedef __attribute__((ext_vector_type(2)))  int    i32x2;
typedef __attribute__((ext_vector_type(4)))  int    i32x4;
typedef __attribute__((ext_vector_type(8)))  int    i32x8;
typedef __attribute__((ext_vector_type(4)))  float  f32x4;
typedef __attribute__((ext_vector_type(8)))  float  f32x8;
typedef __attribute__((ext_vector_type(16))) __bf16 bf16x16;

constexpr int KSTEP = 32;   // K depth of one WMMA bf16 op
constexpr int LDSS  = 40;   // padded LDS row stride in bf16 elems (80 B, 16B-aligned)

#if defined(__HIP_DEVICE_COMPILE__) && \
    __has_builtin(__builtin_amdgcn_global_load_async_to_lds_b128)
#define HAVE_ASYNC_LDS 1
#else
#define HAVE_ASYNC_LDS 0
#endif

__device__ __forceinline__ unsigned short f2b(float f) {
  unsigned int u = __float_as_uint(f);
  return (unsigned short)((u + 0x7FFFu + ((u >> 16) & 1u)) >> 16);  // RNE
}
__device__ __forceinline__ int pack2(float a, float b) {
  return (int)((unsigned int)f2b(a) | ((unsigned int)f2b(b) << 16));
}

#if HAVE_ASYNC_LDS
// Address-space-qualified vector types matching the async builtin signature:
// (v4i addrspace(1)* src, v4i addrspace(3)* dst, Imm offset, Imm cpol).
typedef __attribute__((address_space(1))) i32x4 gbl_i32x4;
typedef __attribute__((address_space(3))) i32x4 lds_i32x4;

__device__ __forceinline__ void wait_async() {
#if __has_builtin(__builtin_amdgcn_s_wait_asynccnt)
  __builtin_amdgcn_s_wait_asynccnt(0);
#else
  asm volatile("s_wait_asynccnt 0" ::: "memory");
#endif
}
// DMA one BLK x 32 bf16 tile (row-major, leading dim ldK) into LDS.
// Each lane moves 16B; ASYNCcnt-tracked, no VGPR round trip.
template <int BLK>
__device__ __forceinline__ void async_tile_bf16(unsigned short* dst,
                                                const unsigned short* src,
                                                int ldK, int tid) {
  constexpr int ITER = BLK / 64;            // (BLK*4 items) / 256 threads
  #pragma unroll
  for (int it = 0; it < ITER; ++it) {
    const int i = tid + it * 256;
    const int row = i >> 2, seg = i & 3;
    __builtin_amdgcn_global_load_async_to_lds_b128(
        (gbl_i32x4*)(void*)(src + (long long)row * ldK + seg * 8),
        (lds_i32x4*)(void*)(dst + row * LDSS + seg * 8),
        0, 0);
  }
}
#endif  // HAVE_ASYNC_LDS

// ---- synchronous fallback loaders (loads batched before LDS stores) ----
template <int BLK>
__device__ __forceinline__ void stage_sync_bf16(unsigned short* dst,
                                                const unsigned short* __restrict__ src,
                                                int ldK, int tid) {
  constexpr int ITER = BLK / 64;
  i32x4 tmp[ITER];
  #pragma unroll
  for (int it = 0; it < ITER; ++it) {
    const int i = tid + it * 256;
    const int row = i >> 2, seg = i & 3;
    tmp[it] = *(const i32x4*)(src + (long long)row * ldK + seg * 8);
  }
  #pragma unroll
  for (int it = 0; it < ITER; ++it) {
    const int i = tid + it * 256;
    const int row = i >> 2, seg = i & 3;
    *(i32x4*)(dst + row * LDSS + seg * 8) = tmp[it];
  }
}
template <int BLK>
__device__ __forceinline__ void stage_sync_f32(unsigned short* dst,
                                               const float* __restrict__ src,
                                               int ldK, int tid) {
  constexpr int ITER = BLK / 64;
  f32x4 ta[ITER], tb[ITER];
  #pragma unroll
  for (int it = 0; it < ITER; ++it) {
    const int i = tid + it * 256;
    const int row = i >> 2, seg = i & 3;
    const float* p = src + (long long)row * ldK + seg * 8;
    ta[it] = *(const f32x4*)p;
    tb[it] = *(const f32x4*)(p + 4);
  }
  #pragma unroll
  for (int it = 0; it < ITER; ++it) {
    const int i = tid + it * 256;
    const int row = i >> 2, seg = i & 3;
    i32x4 d;
    d[0] = pack2(ta[it][0], ta[it][1]); d[1] = pack2(ta[it][2], ta[it][3]);
    d[2] = pack2(tb[it][0], tb[it][1]); d[3] = pack2(tb[it][2], tb[it][3]);
    *(i32x4*)(dst + row * LDSS + seg * 8) = d;
  }
}

// ---- WMMA fragment builders (CDNA5 ISA 7.12.2 layouts, wave32) ----
// A (16x32 bf16): lane m = row0+(lane&15); lane[4] selects K-half:
//   v0..v3 = K [hi*8 .. +7], v4..v7 = K [16+hi*8 .. +7]  -> 2x ds_read_b128
__device__ __forceinline__ bf16x16 frag_a(const unsigned short* As, int row0, int lane) {
  const int r  = row0 + (lane & 15);
  const int hi = lane >> 4;
  const unsigned short* p = As + r * LDSS;
  i32x4 lo = *(const i32x4*)(p + hi * 8);
  i32x4 hq = *(const i32x4*)(p + 16 + hi * 8);
  i32x8 w;
  w[0]=lo[0]; w[1]=lo[1]; w[2]=lo[2]; w[3]=lo[3];
  w[4]=hq[0]; w[5]=hq[1]; w[6]=hq[2]; w[7]=hq[3];
  return __builtin_bit_cast(bf16x16, w);
}
// B (32x16 bf16), K-contiguous per N row: lane n = col0+(lane&15);
//   v0..v3 = K [hi*8 .. +7], v4..v7 = K [16+hi*8 .. +7]  -> 2x ds_read_b128
__device__ __forceinline__ bf16x16 frag_b(const unsigned short* Bs, int col0, int lane) {
  const int c  = col0 + (lane & 15);
  const int hi = lane >> 4;
  const unsigned short* p = Bs + c * LDSS;
  i32x4 lo = *(const i32x4*)(p + hi * 8);
  i32x4 hq = *(const i32x4*)(p + 16 + hi * 8);
  i32x8 w;
  w[0]=lo[0]; w[1]=lo[1]; w[2]=lo[2]; w[3]=lo[3];
  w[4]=hq[0]; w[5]=hq[1]; w[6]=hq[2]; w[7]=hq[3];
  return __builtin_bit_cast(bf16x16, w);
}

// ---- epilogue variants ----
enum {
  EPI_SPLIT_QK = 0,      // +bias, bf16 out to [B,H,S,DH]
  EPI_SPLIT_VT,          // +bias, bf16 out to [B,H,DH,S]
  EPI_SCORES,            // *scale, causal mask, f32 out to attn[B*H, S, S]
  EPI_CTX,               // bf16 out to [B,S,D] at column h*64+n
  EPI_BIAS_RES_F32,      // +bias +residual, f32 out row-major
  EPI_BIAS_RELU_BF16     // relu(+bias), bf16 out row-major
};

struct EpiParams {
  float*          outF;
  unsigned short* outB;
  const float*    bias;
  const float*    resid;
  float           scale;
};

// ---------------------------------------------------------------------------
// NT GEMM: C[M,N] = A[M,K] * Bt[N,K]^T, 256 threads = 8 waves.
// Wave grid WROWS x WCOLS; each wave computes (WT_M/16)x(WT_N/16) WMMA tiles.
// Async path: double-buffered LDS, DMA of tile k+1 overlaps WMMA on tile k.
// ---------------------------------------------------------------------------
template <int M_BLK, int N_BLK, int WROWS, int WCOLS, int EPI, bool A_F32, bool CAUSAL_K>
__global__ __launch_bounds__(256) void gemm_nt(
    const void* __restrict__ Ap, const unsigned short* __restrict__ Btp,
    EpiParams ep, int M, int N, int K,
    long long strideA, long long strideB)
{
  constexpr int WT_M = M_BLK / WROWS;
  constexpr int WT_N = N_BLK / WCOLS;
  constexpr int SM   = WT_M / 16;
  constexpr int SN   = WT_N / 16;

  __shared__ __align__(16) unsigned short As[2][M_BLK * LDSS];
  __shared__ __align__(16) unsigned short Bs[2][N_BLK * LDSS];

  const int tid   = threadIdx.x;
  const int lane  = tid & 31;
  const int wv    = tid >> 5;
  const int wRow  = wv / WCOLS;
  const int wCol  = wv % WCOLS;
  const int mBase = (int)blockIdx.y * M_BLK;
  const int nBase = (int)blockIdx.x * N_BLK;
  const int bh    = (int)blockIdx.z;

  if constexpr (EPI == EPI_SCORES) {
    // Block fully above the causal diagonal: just fill -1e9 (softmax -> 0).
    if (blockIdx.x > blockIdx.y) {
      float* out = ep.outF + (long long)bh * (long long)M * N;
      f32x4 neg; neg[0] = neg[1] = neg[2] = neg[3] = -1e9f;
      for (int i = tid; i < M_BLK * (N_BLK / 4); i += 256) {
        const int r = i / (N_BLK / 4);
        const int c = (i % (N_BLK / 4)) * 4;
        *(f32x4*)(out + (long long)(mBase + r) * N + (nBase + c)) = neg;
      }
      return;
    }
  }

  const long long aOff = (long long)bh * strideA + (long long)mBase * K;
  const long long bOff = (long long)bh * strideB + (long long)nBase * K;

  int kEnd = K;
  if constexpr (CAUSAL_K) {
    const int kc = ((int)blockIdx.y + 1) * M_BLK;   // attn row m has cols <= m
    kEnd = kc < K ? kc : K;
  }
  const int nsteps = kEnd / KSTEP;

  f32x8 acc[SM][SN];
  #pragma unroll
  for (int i = 0; i < SM; ++i)
    #pragma unroll
    for (int j = 0; j < SN; ++j)
      #pragma unroll
      for (int r = 0; r < 8; ++r) acc[i][j][r] = 0.0f;

  auto mma_step = [&](const unsigned short* Acur, const unsigned short* Bcur) {
    bf16x16 fa[SM], fb[SN];
    #pragma unroll
    for (int i = 0; i < SM; ++i) fa[i] = frag_a(Acur, wRow * WT_M + i * 16, lane);
    #pragma unroll
    for (int j = 0; j < SN; ++j) fb[j] = frag_b(Bcur, wCol * WT_N + j * 16, lane);
    #pragma unroll
    for (int i = 0; i < SM; ++i)
      #pragma unroll
      for (int j = 0; j < SN; ++j)
        acc[i][j] = __builtin_amdgcn_wmma_f32_16x16x32_bf16(
            false, fa[i], false, fb[j], (short)0, acc[i][j], false, false);
  };

#if HAVE_ASYNC_LDS
  constexpr bool USE_ASYNC = !A_F32;
#else
  constexpr bool USE_ASYNC = false;
#endif

  if constexpr (USE_ASYNC) {
#if HAVE_ASYNC_LDS
    const unsigned short* Abf = (const unsigned short*)Ap;
    async_tile_bf16<M_BLK>(As[0], Abf + aOff, K, tid);
    async_tile_bf16<N_BLK>(Bs[0], Btp + bOff, K, tid);
    wait_async();
    __syncthreads();
    for (int st = 0; st < nsteps; ++st) {
      const int buf = st & 1;
      if (st + 1 < nsteps) {  // DMA next K tile into the alternate buffer
        const long long k1 = (long long)(st + 1) * KSTEP;
        async_tile_bf16<M_BLK>(As[buf ^ 1], Abf + aOff + k1, K, tid);
        async_tile_bf16<N_BLK>(Bs[buf ^ 1], Btp + bOff + k1, K, tid);
      }
      mma_step(As[buf], Bs[buf]);
      wait_async();
      __syncthreads();
    }
#endif
  } else {
    for (int st = 0; st < nsteps; ++st) {
      const long long k0 = (long long)st * KSTEP;
      if constexpr (A_F32)
        stage_sync_f32<M_BLK>(As[0], (const float*)Ap + aOff + k0, K, tid);
      else
        stage_sync_bf16<M_BLK>(As[0], (const unsigned short*)Ap + aOff + k0, K, tid);
      stage_sync_bf16<N_BLK>(Bs[0], Btp + bOff + k0, K, tid);
      if (st + 1 < nsteps) {  // L2 hint for the next K tile (global_prefetch)
        const long long esz = A_F32 ? 4 : 2;
        __builtin_prefetch((const char*)Ap +
            (aOff + (long long)(tid & (M_BLK - 1)) * K + k0 + KSTEP) * esz, 0, 1);
        __builtin_prefetch((const char*)(Btp + bOff +
            (long long)(tid & (N_BLK - 1)) * K + k0 + KSTEP), 0, 1);
      }
      __syncthreads();
      mma_step(As[0], Bs[0]);
      __syncthreads();
    }
  }

  // ---- epilogue: C layout is VGPR r <-> M = r + 8*lane[4], N = lane[3:0] ----
  const int hi = lane >> 4;
  const int ln = lane & 15;
  #pragma unroll
  for (int i = 0; i < SM; ++i) {
    #pragma unroll
    for (int j = 0; j < SN; ++j) {
      #pragma unroll
      for (int r = 0; r < 8; ++r) {
        const int m = mBase + wRow * WT_M + i * 16 + hi * 8 + r;
        const int n = nBase + wCol * WT_N + j * 16 + ln;
        float c = acc[i][j][r];
        if constexpr (EPI == EPI_SPLIT_QK) {
          c += ep.bias[n];
          const int b = m >> 10, s = m & 1023, h = n >> 6, dh = n & 63;
          ep.outB[(((long long)(b * 16 + h)) * 1024 + s) * 64 + dh] = f2b(c);
        } else if constexpr (EPI == EPI_SPLIT_VT) {
          c += ep.bias[n];
          const int b = m >> 10, s = m & 1023, h = n >> 6, dh = n & 63;
          ep.outB[(((long long)(b * 16 + h)) * 64 + dh) * 1024 + s] = f2b(c);
        } else if constexpr (EPI == EPI_SCORES) {
          c *= ep.scale;
          if (n > m) c = -1e9f;
          ep.outF[(long long)bh * M * N + (long long)m * N + n] = c;
        } else if constexpr (EPI == EPI_CTX) {
          const int b = bh >> 4, h = bh & 15;
          ep.outB[((long long)b * 1024 + m) * 1024 + h * 64 + n] = f2b(c);
        } else if constexpr (EPI == EPI_BIAS_RES_F32) {
          c += ep.bias[n] + ep.resid[(long long)m * N + n];
          ep.outF[(long long)m * N + n] = c;
        } else if constexpr (EPI == EPI_BIAS_RELU_BF16) {
          c += ep.bias[n];
          ep.outB[(long long)m * N + n] = f2b(c > 0.0f ? c : 0.0f);
        }
      }
    }
  }
}

// ---------------------------------------------------------------------------
// Row softmax over S=1024 (in place), one block per row.
// ---------------------------------------------------------------------------
__global__ __launch_bounds__(256) void softmax_rows(float* __restrict__ attn) {
  constexpr int S = 1024;
  float* p = attn + (long long)blockIdx.x * S;
  const int tid = threadIdx.x;
  __shared__ float red[256];

  f32x4 x = *(const f32x4*)(p + tid * 4);
  float mx = fmaxf(fmaxf(x[0], x[1]), fmaxf(x[2], x[3]));
  red[tid] = mx; __syncthreads();
  #pragma unroll
  for (int s = 128; s > 0; s >>= 1) {
    if (tid < s) red[tid] = fmaxf(red[tid], red[tid + s]);
    __syncthreads();
  }
  mx = red[0]; __syncthreads();

  x[0] = __expf(x[0] - mx); x[1] = __expf(x[1] - mx);
  x[2] = __expf(x[2] - mx); x[3] = __expf(x[3] - mx);
  red[tid] = x[0] + x[1] + x[2] + x[3]; __syncthreads();
  #pragma unroll
  for (int s = 128; s > 0; s >>= 1) {
    if (tid < s) red[tid] += red[tid + s];
    __syncthreads();
  }
  const float inv = 1.0f / red[0];
  x[0] *= inv; x[1] *= inv; x[2] *= inv; x[3] *= inv;
  *(f32x4*)(p + tid * 4) = x;
}

// ---------------------------------------------------------------------------
// LayerNorm over D=1024, one block per row. Optionally emits bf16 copy.
// ---------------------------------------------------------------------------
template <bool WRITE_BF16>
__global__ __launch_bounds__(256) void layernorm_rows(
    const float* __restrict__ in, const float* __restrict__ g,
    const float* __restrict__ be, float* __restrict__ outF,
    unsigned short* __restrict__ outB) {
  constexpr int D = 1024;
  const long long row = blockIdx.x;
  const int tid = threadIdx.x;
  __shared__ float red[256];

  f32x4 x = *(const f32x4*)(in + row * D + tid * 4);
  red[tid] = x[0] + x[1] + x[2] + x[3]; __syncthreads();
  #pragma unroll
  for (int s = 128; s > 0; s >>= 1) {
    if (tid < s) red[tid] += red[tid + s];
    __syncthreads();
  }
  const float mean = red[0] * (1.0f / D); __syncthreads();

  const float d0 = x[0] - mean, d1 = x[1] - mean, d2 = x[2] - mean, d3 = x[3] - mean;
  red[tid] = d0 * d0 + d1 * d1 + d2 * d2 + d3 * d3; __syncthreads();
  #pragma unroll
  for (int s = 128; s > 0; s >>= 1) {
    if (tid < s) red[tid] += red[tid + s];
    __syncthreads();
  }
  const float rs = rsqrtf(red[0] * (1.0f / D) + 1e-5f);

  f32x4 gg = *(const f32x4*)(g + tid * 4);
  f32x4 bb = *(const f32x4*)(be + tid * 4);
  f32x4 y;
  y[0] = d0 * rs * gg[0] + bb[0];
  y[1] = d1 * rs * gg[1] + bb[1];
  y[2] = d2 * rs * gg[2] + bb[2];
  y[3] = d3 * rs * gg[3] + bb[3];
  *(f32x4*)(outF + row * D + tid * 4) = y;
  if constexpr (WRITE_BF16) {
    i32x2 d; d[0] = pack2(y[0], y[1]); d[1] = pack2(y[2], y[3]);
    *(i32x2*)(outB + row * D + tid * 4) = d;
  }
}

// ---------------------------------------------------------------------------
// f32 -> bf16 straight conversion.
// ---------------------------------------------------------------------------
__global__ __launch_bounds__(256) void convert_f32_to_bf16(
    const float* __restrict__ in, unsigned short* __restrict__ out, long long n) {
  const long long i = ((long long)blockIdx.x * 256 + threadIdx.x) * 4;
  if (i + 3 < n) {
    f32x4 x = *(const f32x4*)(in + i);
    i32x2 d; d[0] = pack2(x[0], x[1]); d[1] = pack2(x[2], x[3]);
    *(i32x2*)(out + i) = d;
  }
}

// ---------------------------------------------------------------------------
// Tiled transpose + convert: W[R][C] f32 -> Wt[C][R] bf16 (K-contiguous).
// ---------------------------------------------------------------------------
__global__ __launch_bounds__(256) void transpose_f32_to_bf16(
    const float* __restrict__ in, unsigned short* __restrict__ out, int R, int C) {
  __shared__ float tile[32][33];
  const int tx = threadIdx.x, ty = threadIdx.y;       // 32 x 8
  const int c0 = blockIdx.x * 32, r0 = blockIdx.y * 32;
  #pragma unroll
  for (int i = ty; i < 32; i += 8)
    tile[i][tx] = in[(long long)(r0 + i) * C + (c0 + tx)];
  __syncthreads();
  #pragma unroll
  for (int i = ty; i < 32; i += 8)
    out[(long long)(c0 + i) * R + (r0 + tx)] = f2b(tile[tx][i]);
}

// ---------------------------------------------------------------------------
// Host-side orchestration.
// ---------------------------------------------------------------------------
extern "C" void kernel_launch(void* const* d_in, const int* in_sizes, int n_in,
                              void* d_out, int out_size, void* d_ws, size_t ws_size,
                              hipStream_t stream) {
  (void)in_sizes; (void)n_in; (void)out_size; (void)ws_size;
  constexpr int B = 4, S = 1024, D = 1024, H = 16, DH = 64, FF = 4096;
  constexpr long long BS = (long long)B * S;            // 4096 tokens

  const float* x    = (const float*)d_in[0];
  const float* Wq   = (const float*)d_in[1];
  const float* bq   = (const float*)d_in[2];
  const float* Wk   = (const float*)d_in[3];
  const float* bk   = (const float*)d_in[4];
  const float* Wv   = (const float*)d_in[5];
  const float* bv   = (const float*)d_in[6];
  const float* Wo   = (const float*)d_in[7];
  const float* bo   = (const float*)d_in[8];
  const float* ln1g = (const float*)d_in[9];
  const float* ln1b = (const float*)d_in[10];
  const float* W1   = (const float*)d_in[11];
  const float* b1   = (const float*)d_in[12];
  const float* W2   = (const float*)d_in[13];
  const float* b2   = (const float*)d_in[14];
  const float* ln2g = (const float*)d_in[15];
  const float* ln2b = (const float*)d_in[16];

  float* y    = (float*)d_out;                          // [B,S,D]
  float* attn = y + BS * D;                             // [B,H,S,S]

  // ---- carve workspace (aligned 256B); total ~152 MB ----
  char* wsp = (char*)d_ws;
  auto carve = [&](size_t bytes) {
    char* p = wsp; wsp += (bytes + 255) & ~(size_t)255; return (void*)p;
  };
  unsigned short* xb   = (unsigned short*)carve(BS * D * 2);
  unsigned short* Wqt  = (unsigned short*)carve((size_t)D * D * 2);
  unsigned short* Wkt  = (unsigned short*)carve((size_t)D * D * 2);
  unsigned short* Wvt  = (unsigned short*)carve((size_t)D * D * 2);
  unsigned short* Wot  = (unsigned short*)carve((size_t)D * D * 2);
  unsigned short* W1t  = (unsigned short*)carve((size_t)D * FF * 2);  // [FF][D]
  unsigned short* W2t  = (unsigned short*)carve((size_t)D * FF * 2);  // [D][FF]
  unsigned short* qb   = (unsigned short*)carve(BS * D * 2);          // [B,H,S,DH]
  unsigned short* kb   = (unsigned short*)carve(BS * D * 2);          // [B,H,S,DH]
  unsigned short* vtb  = (unsigned short*)carve(BS * D * 2);          // [B,H,DH,S]
  unsigned short* ctxb = (unsigned short*)carve(BS * D * 2);          // [B,S,D]
  float*          res1 = (float*)carve(BS * D * 4);
  float*          hF   = (float*)carve(BS * D * 4);
  unsigned short* hB   = (unsigned short*)carve(BS * D * 2);
  unsigned short* ff1b = (unsigned short*)carve(BS * FF * 2);
  float*          res2 = (float*)carve(BS * D * 4);

  const dim3 t256(256);
  const dim3 tT(32, 8);

  // ---- precision conversion / weight transposition (bf16, K-contiguous) ----
  convert_f32_to_bf16<<<(unsigned)(BS * D / 1024), t256, 0, stream>>>(x, xb, BS * D);
  transpose_f32_to_bf16<<<dim3(D / 32, D / 32), tT, 0, stream>>>(Wq, Wqt, D, D);
  transpose_f32_to_bf16<<<dim3(D / 32, D / 32), tT, 0, stream>>>(Wk, Wkt, D, D);
  transpose_f32_to_bf16<<<dim3(D / 32, D / 32), tT, 0, stream>>>(Wv, Wvt, D, D);
  transpose_f32_to_bf16<<<dim3(D / 32, D / 32), tT, 0, stream>>>(Wo, Wot, D, D);
  transpose_f32_to_bf16<<<dim3(FF / 32, D / 32), tT, 0, stream>>>(W1, W1t, D, FF);
  transpose_f32_to_bf16<<<dim3(D / 32, FF / 32), tT, 0, stream>>>(W2, W2t, FF, D);

  EpiParams ep{};

  // ---- QKV projections: [4096,1024] = xb @ W*, split-head epilogues ----
  ep = {}; ep.outB = qb; ep.bias = bq;
  gemm_nt<128, 128, 2, 4, EPI_SPLIT_QK, false, false>
      <<<dim3(D / 128, BS / 128, 1), t256, 0, stream>>>(xb, Wqt, ep, (int)BS, D, D, 0, 0);
  ep = {}; ep.outB = kb; ep.bias = bk;
  gemm_nt<128, 128, 2, 4, EPI_SPLIT_QK, false, false>
      <<<dim3(D / 128, BS / 128, 1), t256, 0, stream>>>(xb, Wkt, ep, (int)BS, D, D, 0, 0);
  ep = {}; ep.outB = vtb; ep.bias = bv;
  gemm_nt<128, 128, 2, 4, EPI_SPLIT_VT, false, false>
      <<<dim3(D / 128, BS / 128, 1), t256, 0, stream>>>(xb, Wvt, ep, (int)BS, D, D, 0, 0);

  // ---- scores = Q K^T / 8, causal mask; batched over B*H ----
  ep = {}; ep.outF = attn; ep.scale = 0.125f;
  gemm_nt<128, 128, 2, 4, EPI_SCORES, false, false>
      <<<dim3(S / 128, S / 128, B * H), t256, 0, stream>>>(
          qb, kb, ep, S, S, DH, (long long)S * DH, (long long)S * DH);

  // ---- softmax in place over attn rows ----
  softmax_rows<<<(unsigned)(B * H * S), t256, 0, stream>>>(attn);

  // ---- ctx = attn @ V  (A f32->bf16 on the fly, causal K bound) ----
  ep = {}; ep.outB = ctxb;
  gemm_nt<128, 64, 4, 2, EPI_CTX, true, true>
      <<<dim3(1, S / 128, B * H), t256, 0, stream>>>(
          attn, vtb, ep, S, DH, S, (long long)S * S, (long long)DH * S);

  // ---- attn_out = ctx @ Wo + bo + x  ->  res1 ----
  ep = {}; ep.outF = res1; ep.bias = bo; ep.resid = x;
  gemm_nt<128, 128, 2, 4, EPI_BIAS_RES_F32, false, false>
      <<<dim3(D / 128, BS / 128, 1), t256, 0, stream>>>(ctxb, Wot, ep, (int)BS, D, D, 0, 0);

  // ---- h = LN1(res1) (f32 + bf16 copies) ----
  layernorm_rows<true><<<(unsigned)BS, t256, 0, stream>>>(res1, ln1g, ln1b, hF, hB);

  // ---- ff1 = relu(h @ W1 + b1) (bf16) ----
  ep = {}; ep.outB = ff1b; ep.bias = b1;
  gemm_nt<128, 128, 2, 4, EPI_BIAS_RELU_BF16, false, false>
      <<<dim3(FF / 128, BS / 128, 1), t256, 0, stream>>>(hB, W1t, ep, (int)BS, FF, D, 0, 0);

  // ---- res2 = ff1 @ W2 + b2 + h ----
  ep = {}; ep.outF = res2; ep.bias = b2; ep.resid = hF;
  gemm_nt<128, 128, 2, 4, EPI_BIAS_RES_F32, false, false>
      <<<dim3(D / 128, BS / 128, 1), t256, 0, stream>>>(ff1b, W2t, ep, (int)BS, D, FF, 0, 0);

  // ---- y = LN2(res2) ----
  layernorm_rows<false><<<(unsigned)BS, t256, 0, stream>>>(res2, ln2g, ln2b, y, nullptr);
}
